// EfficientAttention_19404662243642
// MI455X (gfx1250) — matrix operand order repaired and checked by reading, not compiled, not run on hardware
//
#include <hip/hip_runtime.h>
#include <stdint.h>

// ---------------------------------------------------------------------------
// EfficientAttention for MI455X (gfx1250, wave32, WMMA bf16 16x16x32).
//
// x:[8,256,128,128] f32.  3x 1x1-conv GEMMs + linear attention + residual.
// Roofline: ~56 GFLOP vs >=1.5GB mandatory HBM traffic -> memory bound
// (~65us floor @23.3TB/s).  So: never materialize K/Q/V in HBM.
//   prep      : W{k,q,v} f32 -> bf16 in workspace (L2-resident afterwards)
//   kv_ctx    : per (n, 512-pixel slab): read x once (float4, prefetch next
//               subtile), WMMA K and V tiles in LDS, exp(k) (softmax over L is
//               shift invariant; k ~ N(0,0.32) so no max-subtraction needed),
//               accumulate per-slab ctx partials (32x32/head via WMMA over l)
//               and per-channel exp-row-sums.  Deterministic partials to ws.
//   reduce_ctx: fold slab partials, divide by row sums, emit ctx^T bf16.
//   q_att     : recompute Q from x, 32-channel softmax in LDS, att = ctx^T q
//               (one WMMA step), bounce att through LDS so the residual
//               read-modify-write is fully coalesced float4 traffic.
// ---------------------------------------------------------------------------

#define C_DIM   256
#define L_DIM   16384
#define HEADS   8
#define N_IMG   8

#define NTHREADS 512          // 16 waves of 32
#define NW       16

// kernel 2 tiling
#define K2_LT    64           // l subtile
#define K2_SUBT  8            // subtiles per slab
#define SLAB     (K2_LT * K2_SUBT)   // 512
#define SLABS    (L_DIM / SLAB)      // 32
#define KV_PITCH 72           // (64+8) halves; 144B rows -> 16B aligned
// kernel 4 tiling
#define K4_LT    128

#define XT_PITCH 264          // (256+8) halves; 528B rows -> 16B aligned
#define AF_PITCH 132          // floats; 528B rows -> 16B aligned

typedef __bf16 bf16x16 __attribute__((ext_vector_type(16)));
typedef float  f32x8   __attribute__((ext_vector_type(8)));
typedef unsigned int u32x4 __attribute__((ext_vector_type(4)));

union Frag { bf16x16 v; u32x4 q[2]; };

__device__ inline unsigned short toBF(float x) {
  unsigned int u = __float_as_uint(x);
  u += 0x7FFFu + ((u >> 16) & 1u);          // round-to-nearest-even
  return (unsigned short)(u >> 16);
}
__device__ inline float fromBF(unsigned short s) {
  return __uint_as_float(((unsigned int)s) << 16);
}

// A fragment: 16x32 bf16, row-major, K contiguous.  lane m<16 holds row m,
// K chunks {0..7,16..23}; lane m+16 holds K chunks {8..15,24..31}. (ISA 7.12.2)
__device__ inline bf16x16 loadA(const unsigned short* base, int rowStride, int lane) {
  int r = lane & 15, hi = (lane >> 4) & 1;
  const unsigned short* p = base + r * rowStride + hi * 8;
  Frag f;
  f.q[0] = *(const u32x4*)(p);
  f.q[1] = *(const u32x4*)(p + 16);
  return f.v;
}
// B fragment: 32x16 bf16 stored as 16 rows [n][k], K contiguous.  lane n<16
// holds K {0..7,16..23} of column n; lane n+16 holds K {8..15,24..31}.
__device__ inline bf16x16 loadB(const unsigned short* base, int rowStride, int lane) {
  int n = lane & 15, hi = (lane >> 4) & 1;
  const unsigned short* p = base + n * rowStride + hi * 8;
  Frag f;
  f.q[0] = *(const u32x4*)(p);
  f.q[1] = *(const u32x4*)(p + 16);
  return f.v;
}

__device__ inline f32x8 wmma_bf16(bf16x16 a, bf16x16 b, f32x8 c) {
  return __builtin_amdgcn_wmma_f32_16x16x32_bf16(false, a, false, b,
                                                 (short)0, c, false, false);
}

// ---------------------------------------------------------------------------
__global__ void prep_weights(const float* __restrict__ Wk, const float* __restrict__ Wq,
                             const float* __restrict__ Wv,
                             unsigned short* __restrict__ WkB,
                             unsigned short* __restrict__ WqB,
                             unsigned short* __restrict__ WvB) {
  int i = blockIdx.x * blockDim.x + threadIdx.x;
  if (i < C_DIM * C_DIM) {
    WkB[i] = toBF(Wk[i]);
    WqB[i] = toBF(Wq[i]);
    WvB[i] = toBF(Wv[i]);
  }
}

// ---------------------------------------------------------------------------
// One workgroup per (slab, n).  16 waves.  Wave w owns output-channel rows
// [16w,16w+16) for the projection GEMMs, and 2 of the 32 (head,quadrant)
// ctx tiles.  LDS = 107.6KB -> two workgroups co-resident per 320KB WGP.
__global__ __launch_bounds__(NTHREADS)
void kv_ctx(const float* __restrict__ x,
            const unsigned short* __restrict__ Wk,
            const unsigned short* __restrict__ Wv,
            const float* __restrict__ bk, const float* __restrict__ bv,
            float* __restrict__ ctxPart, float* __restrict__ ksumPart) {
  __shared__ __align__(16) unsigned short xT[K2_LT * XT_PITCH];   // [l][cin]
  __shared__ __align__(16) unsigned short eK[C_DIM * KV_PITCH];   // [c][l] exp(k)
  __shared__ __align__(16) unsigned short vS[C_DIM * KV_PITCH];   // [c][l] v

  const int slab = blockIdx.x;
  const int n    = blockIdx.y;
  const int tid  = threadIdx.x;
  const int lane = tid & 31;
  const int wv   = tid >> 5;
  const int hi   = (lane >> 4) & 1;
  const int nn   = lane & 15;
  const int c0   = wv * 16;

  const float* xn = x + (size_t)n * C_DIM * L_DIM;

  float bkv[8], bvv[8];
#pragma unroll
  for (int e = 0; e < 8; ++e) {
    bkv[e] = bk[c0 + hi * 8 + e];
    bvv[e] = bv[c0 + hi * 8 + e];
  }

  // ctx accumulators: wave w -> tiles 2w, 2w+1 of 32 (8 heads x 2x2 quads)
  f32x8 ctxA[2];
  ctxA[0] = {};
  ctxA[1] = {};
  float ksum_r = 0.f;   // thread t<256 accumulates channel t

  for (int s = 0; s < K2_SUBT; ++s) {
    const int l0 = slab * SLAB + s * K2_LT;
    __syncthreads();                       // prior iter done with LDS
    // ---- stage x tile, transposed, bf16 (float4 -> global_load_b128) ----
    for (int i = tid; i < C_DIM * (K2_LT / 4); i += NTHREADS) {
      int c = i >> 4, lc = (i & 15) << 2;
      const float4 v4 = *(const float4*)(xn + (size_t)c * L_DIM + l0 + lc);
      unsigned short* p = xT + lc * XT_PITCH + c;
      p[0 * XT_PITCH] = toBF(v4.x);
      p[1 * XT_PITCH] = toBF(v4.y);
      p[2 * XT_PITCH] = toBF(v4.z);
      p[3 * XT_PITCH] = toBF(v4.w);
    }
    // ---- prefetch next subtile into cache while this one computes ----
    if (s + 1 < K2_SUBT) {
      const float* pf = xn + (size_t)(tid >> 1) * L_DIM + (l0 + K2_LT) + (tid & 1) * 32;
      __builtin_prefetch(pf, 0, 0);        // -> global_prefetch_b8
    }
    __syncthreads();

    // ---- K projection: D[c_out, l] = sum_cin Wk * x ----
    f32x8 acc[4];
#pragma unroll
    for (int j = 0; j < 4; ++j) acc[j] = {};
    for (int kk = 0; kk < C_DIM; kk += 32) {
      bf16x16 a = loadA(Wk + c0 * C_DIM + kk, C_DIM, lane);
#pragma unroll
      for (int j = 0; j < 4; ++j) {
        bf16x16 b = loadB(xT + j * 16 * XT_PITCH + kk, XT_PITCH, lane);
        acc[j] = wmma_bf16(a, b, acc[j]);
      }
    }
#pragma unroll
    for (int j = 0; j < 4; ++j)
#pragma unroll
      for (int e = 0; e < 8; ++e) {
        float kv = __expf(acc[j][e] + bkv[e]);
        eK[(c0 + hi * 8 + e) * KV_PITCH + j * 16 + nn] = toBF(kv);
      }

    // ---- V projection ----
#pragma unroll
    for (int j = 0; j < 4; ++j) acc[j] = {};
    for (int kk = 0; kk < C_DIM; kk += 32) {
      bf16x16 a = loadA(Wv + c0 * C_DIM + kk, C_DIM, lane);
#pragma unroll
      for (int j = 0; j < 4; ++j) {
        bf16x16 b = loadB(xT + j * 16 * XT_PITCH + kk, XT_PITCH, lane);
        acc[j] = wmma_bf16(a, b, acc[j]);
      }
    }
#pragma unroll
    for (int j = 0; j < 4; ++j)
#pragma unroll
      for (int e = 0; e < 8; ++e)
        vS[(c0 + hi * 8 + e) * KV_PITCH + j * 16 + nn] = toBF(acc[j][e] + bvv[e]);
    __syncthreads();

    // ---- exp(k) row-sum partial (softmax denominator over L), vectorized ----
    if (tid < C_DIM) {
      const u32x4* p = (const u32x4*)(eK + tid * KV_PITCH);
      float srow = 0.f;
#pragma unroll
      for (int t = 0; t < K2_LT / 8; ++t) {
        u32x4 q = p[t];
#pragma unroll
        for (int w = 0; w < 4; ++w) {
          unsigned int u = q[w];
          srow += __uint_as_float(u << 16) + __uint_as_float(u & 0xFFFF0000u);
        }
      }
      ksum_r += srow;
    }
    // ---- ctx partial: ctx[ck,cv] += sum_l expK[ck,l] * V[cv,l] ----
#pragma unroll
    for (int t = 0; t < 2; ++t) {
      int tile = wv * 2 + t;
      int h = tile >> 2, quad = tile & 3;
      int ck0 = (quad >> 1) * 16, cv0 = (quad & 1) * 16;
#pragma unroll
      for (int ks = 0; ks < K2_LT; ks += 32) {
        bf16x16 a = loadA(eK + (h * 32 + ck0) * KV_PITCH + ks, KV_PITCH, lane);
        bf16x16 b = loadB(vS + (h * 32 + cv0) * KV_PITCH + ks, KV_PITCH, lane);
        ctxA[t] = wmma_bf16(a, b, ctxA[t]);
      }
    }
  }

  // ---- write per-slab partials (deterministic reduction in next kernel) ----
  if (tid < C_DIM)
    ksumPart[((size_t)n * SLABS + slab) * C_DIM + tid] = ksum_r;
#pragma unroll
  for (int t = 0; t < 2; ++t) {
    int tile = wv * 2 + t;
    int h = tile >> 2, quad = tile & 3;
    int ck0 = (quad >> 1) * 16, cv0 = (quad & 1) * 16;
    float* base = ctxPart + (((size_t)n * SLABS + slab) * HEADS + h) * 1024;
#pragma unroll
    for (int e = 0; e < 8; ++e)
      base[(ck0 + hi * 8 + e) * 32 + cv0 + nn] = ctxA[t][e];
  }
}

// ---------------------------------------------------------------------------
// Fold slab partials; ctxT[n][h][cv][ck] = (sum ctx) / ksum[ck]  (bf16)
__global__ __launch_bounds__(256)
void reduce_ctx(const float* __restrict__ ctxPart, const float* __restrict__ ksumPart,
                unsigned short* __restrict__ ctxT) {
  __shared__ float sK[C_DIM];
  int n = blockIdx.x, tid = threadIdx.x;
  float srow = 0.f;
  for (int sl = 0; sl < SLABS; ++sl)
    srow += ksumPart[((size_t)n * SLABS + sl) * C_DIM + tid];
  sK[tid] = srow;
  __syncthreads();
  for (int i = tid; i < HEADS * 1024; i += 256) {
    int h = i >> 10, r = i & 1023;
    int ck = r >> 5, cv = r & 31;
    float acc = 0.f;
    for (int sl = 0; sl < SLABS; ++sl)
      acc += ctxPart[(((size_t)n * SLABS + sl) * HEADS + h) * 1024 + r];
    float val = acc / sK[h * 32 + ck];
    ctxT[((size_t)(n * HEADS + h)) * 1024 + cv * 32 + ck] = toBF(val);
  }
}

// ---------------------------------------------------------------------------
// One workgroup per (128-pixel tile, n).  Q GEMM -> channel softmax -> att.
// att is bounced through LDS (reusing xT+qT space, exact fit) so that the
// residual read-modify-write runs as fully coalesced float4 traffic.
__global__ __launch_bounds__(NTHREADS)
void q_att(const float* __restrict__ x,
           const unsigned short* __restrict__ Wq, const float* __restrict__ bq,
           const unsigned short* __restrict__ ctxT, float* __restrict__ out) {
  __shared__ __align__(16) unsigned short smem4[2 * K4_LT * XT_PITCH];
  unsigned short* xT = smem4;                        // [l][cin]
  unsigned short* qT = smem4 + K4_LT * XT_PITCH;     // [l][c]
  float* attF = (float*)smem4;                       // [c][l] f32, reuses all

  const int lt  = blockIdx.x;           // 0..127
  const int n   = blockIdx.y;
  const int l0  = lt * K4_LT;
  const int tid = threadIdx.x;
  const int lane = tid & 31;
  const int wv  = tid >> 5;
  const int hi  = (lane >> 4) & 1;
  const int nn  = lane & 15;
  const int c0  = wv * 16;

  const float* xn = x + (size_t)n * C_DIM * L_DIM;
  float* on = out + (size_t)n * C_DIM * L_DIM;

  float bqv[8];
#pragma unroll
  for (int e = 0; e < 8; ++e) bqv[e] = bq[c0 + hi * 8 + e];

  // ---- stage x tile, transposed, bf16 (float4 loads) ----
  for (int i = tid; i < C_DIM * (K4_LT / 4); i += NTHREADS) {
    int c = i >> 5, lc = (i & 31) << 2;
    const float4 v4 = *(const float4*)(xn + (size_t)c * L_DIM + l0 + lc);
    unsigned short* p = xT + lc * XT_PITCH + c;
    p[0 * XT_PITCH] = toBF(v4.x);
    p[1 * XT_PITCH] = toBF(v4.y);
    p[2 * XT_PITCH] = toBF(v4.z);
    p[3 * XT_PITCH] = toBF(v4.w);
  }
  __syncthreads();

  // ---- Q projection ----
  f32x8 acc[8];
#pragma unroll
  for (int j = 0; j < 8; ++j) acc[j] = {};
  for (int kk = 0; kk < C_DIM; kk += 32) {
    bf16x16 a = loadA(Wq + c0 * C_DIM + kk, C_DIM, lane);
#pragma unroll
    for (int j = 0; j < 8; ++j) {
      bf16x16 b = loadB(xT + j * 16 * XT_PITCH + kk, XT_PITCH, lane);
      acc[j] = wmma_bf16(a, b, acc[j]);
    }
  }
#pragma unroll
  for (int j = 0; j < 8; ++j)
#pragma unroll
    for (int e = 0; e < 8; ++e)
      qT[(j * 16 + nn) * XT_PITCH + c0 + hi * 8 + e] = toBF(acc[j][e] + bqv[e]);
  __syncthreads();

  // ---- softmax over the 32 channels of each head, per pixel ----
  for (int i = tid; i < K4_LT * HEADS; i += NTHREADS) {
    int l = i >> 3, h = i & 7;
    unsigned short* row = qT + l * XT_PITCH + h * 32;
    float vals[32], m = -1e30f;
#pragma unroll
    for (int c = 0; c < 32; ++c) { vals[c] = fromBF(row[c]); m = fmaxf(m, vals[c]); }
    float ssum = 0.f;
#pragma unroll
    for (int c = 0; c < 32; ++c) { float e = __expf(vals[c] - m); vals[c] = e; ssum += e; }
    float inv = 1.f / ssum;
#pragma unroll
    for (int c = 0; c < 32; ++c) row[c] = toBF(vals[c] * inv);
  }
  __syncthreads();

  // ---- att[cv,l] = sum_ck ctxT[cv,ck] * qs[ck,l]   (K=32: one WMMA step) ----
  const int h = wv >> 1, cv0 = (wv & 1) * 16;  // c0 == h*32 + cv0
  bf16x16 a = loadA(ctxT + ((size_t)(n * HEADS + h)) * 1024 + cv0 * 32, 32, lane);
  f32x8 zero = {};
  f32x8 dAcc[8];
#pragma unroll
  for (int j = 0; j < 8; ++j) {
    bf16x16 b = loadB(qT + j * 16 * XT_PITCH + h * 32, XT_PITCH, lane);
    dAcc[j] = wmma_bf16(a, b, zero);
  }
  __syncthreads();                        // all waves done reading xT/qT

  // ---- park att in LDS (overlays xT/qT: 256*132*4B == 2*128*264*2B) ----
#pragma unroll
  for (int j = 0; j < 8; ++j)
#pragma unroll
    for (int e = 0; e < 8; ++e)
      attF[(c0 + hi * 8 + e) * AF_PITCH + j * 16 + nn] = dAcc[j][e];
  __syncthreads();

  // ---- coalesced residual epilogue: out = att + x (float4 R/M/W) ----
  for (int i = tid; i < C_DIM * (K4_LT / 4); i += NTHREADS) {
    int c = i >> 5, lc = (i & 31) << 2;
    const float* ap = attF + c * AF_PITCH + lc;
    size_t gidx = (size_t)c * L_DIM + l0 + lc;
    float4 xv = *(const float4*)(xn + gidx);
    float4 o;
    o.x = ap[0] + xv.x;
    o.y = ap[1] + xv.y;
    o.z = ap[2] + xv.z;
    o.w = ap[3] + xv.w;
    *(float4*)(on + gidx) = o;
  }
}

// ---------------------------------------------------------------------------
extern "C" void kernel_launch(void* const* d_in, const int* in_sizes, int n_in,
                              void* d_out, int out_size, void* d_ws, size_t ws_size,
                              hipStream_t stream) {
  (void)in_sizes; (void)n_in; (void)out_size; (void)ws_size;
  const float* x  = (const float*)d_in[0];
  const float* Wk = (const float*)d_in[1];
  const float* bk = (const float*)d_in[2];
  const float* Wq = (const float*)d_in[3];
  const float* bq = (const float*)d_in[4];
  const float* Wv = (const float*)d_in[5];
  const float* bv = (const float*)d_in[6];
  float* out = (float*)d_out;

  char* ws = (char*)d_ws;                       // ~8.75 MB used
  unsigned short* WkB  = (unsigned short*)(ws);            // 128 KB
  unsigned short* WqB  = (unsigned short*)(ws + (131072));
  unsigned short* WvB  = (unsigned short*)(ws + (262144));
  unsigned short* ctxT = (unsigned short*)(ws + (393216)); // 128 KB
  float* ksumPart      = (float*)(ws + (524288));          // 256 KB
  float* ctxPart       = (float*)(ws + (786432));          // 8 MB

  prep_weights<<<(C_DIM * C_DIM + 255) / 256, 256, 0, stream>>>(Wk, Wq, Wv, WkB, WqB, WvB);
  kv_ctx<<<dim3(SLABS, N_IMG), NTHREADS, 0, stream>>>(x, WkB, WvB, bk, bv, ctxPart, ksumPart);
  reduce_ctx<<<N_IMG, 256, 0, stream>>>(ctxPart, ksumPart, ctxT);
  q_att<<<dim3(L_DIM / K4_LT, N_IMG), NTHREADS, 0, stream>>>(x, WqB, bq, ctxT, out);
}
